// MultipleFullGaussian2d_11785390260963
// MI455X (gfx1250) — matrix operand order repaired and checked by reading, not compiled, not run on hardware
//
#include <hip/hip_runtime.h>
#include <hip/hip_bf16.h>
#include <math.h>

// ---------------------------------------------------------------------------
// MultipleFullGaussian2d fused kernel set for gfx1250 (MI455X)
//   B=8, C=256, W=H=64 (S=4096), O=4096, N = B*S = 32768, K = C = 256
//   Pass 1: separable normalized gaussian tables gX[o,64], gY[o,64]
//   Pass 2: bf16 WMMA GEMM (W @ X), double-buffered async global->LDS staging
//           (s_wait_asynccnt pipelining), fused gauss-weighted reduction over S
//           -> partial[o, n_chunk64]   (unique writer per cell, no atomics)
//   Pass 3: out[b,o] = sum of 64 chunk partials
// Workspace: gX (1MB) + gY (1MB) + partial (8MB) = 10 MB of d_ws
// ---------------------------------------------------------------------------

typedef __attribute__((ext_vector_type(16))) __bf16 v16bf;
typedef __attribute__((ext_vector_type(8)))  float  v8f;

#define HAVE_ASYNC_LDS (__has_builtin(__builtin_amdgcn_global_load_async_to_lds_b32) && \
                        __has_builtin(__builtin_amdgcn_global_load_async_to_lds_b128))

#if HAVE_ASYNC_LDS
// Parameter types confirmed by round-1 clang diagnostics + round-2 codegen:
// emits global_load_async_to_lds_b32/b128 (GVS mode w/ scale_offset picked up).
typedef int v4i_gnu __attribute__((vector_size(16)));
typedef __attribute__((address_space(1))) int      g_i32;
typedef __attribute__((address_space(1))) v4i_gnu  g_v4i;
typedef __attribute__((address_space(3))) int      l_i32;
typedef __attribute__((address_space(3))) v4i_gnu  l_v4i;

__device__ __forceinline__ void async_g2l_b32(void* l, const void* g) {
  __builtin_amdgcn_global_load_async_to_lds_b32((g_i32*)g, (l_i32*)l, 0, 0);
}
__device__ __forceinline__ void async_g2l_b128(void* l, const void* g) {
  __builtin_amdgcn_global_load_async_to_lds_b128((g_v4i*)g, (l_v4i*)l, 0, 0);
}
#endif

// Build a v16bf fragment from two contiguous 32-byte fp32 runs in LDS.
// (A-frag: K run [ka..ka+15]; B-frag: K run [kb..kb+15])
__device__ __forceinline__ v16bf make_frag(const float* p0, const float* p1) {
  const float4* q0 = (const float4*)p0;
  const float4* q1 = (const float4*)p1;
  float4 a0 = q0[0], a1 = q0[1];
  float4 b0 = q1[0], b1 = q1[1];
  v16bf f;
  f[0]  = (__bf16)a0.x; f[1]  = (__bf16)a0.y; f[2]  = (__bf16)a0.z; f[3]  = (__bf16)a0.w;
  f[4]  = (__bf16)a1.x; f[5]  = (__bf16)a1.y; f[6]  = (__bf16)a1.z; f[7]  = (__bf16)a1.w;
  f[8]  = (__bf16)b0.x; f[9]  = (__bf16)b0.y; f[10] = (__bf16)b0.z; f[11] = (__bf16)b0.w;
  f[12] = (__bf16)b1.x; f[13] = (__bf16)b1.y; f[14] = (__bf16)b1.z; f[15] = (__bf16)b1.w;
  return f;
}

// ---------------------------------------------------------------------------
// Pass 1: normalized separable gaussian tables. 4096 blocks x 64 threads.
// gauss[o, i*64+j] = gX[o*64+j] * gY[o*64+i]
// ---------------------------------------------------------------------------
__global__ void gauss_tables_kernel(const float* __restrict__ mu,
                                    const float* __restrict__ sigma,
                                    float* __restrict__ gX,
                                    float* __restrict__ gY) {
  const int o = blockIdx.x;
  const int t = threadIdx.x;            // 0..63 = pixel coordinate
  const float mux = (1.0f / (1.0f + __expf(-mu[o * 2 + 0]))) * 64.0f;
  const float muy = (1.0f / (1.0f + __expf(-mu[o * 2 + 1]))) * 64.0f;
  const float sx = __expf(sigma[o * 2 + 0]);
  const float sy = __expf(sigma[o * 2 + 1]);
  const float zx = ((float)t - mux) / sx;
  const float zy = ((float)t - muy) / sy;
  const float gx = __expf(-0.5f * zx * zx);
  const float gy = __expf(-0.5f * zy * zy);
  __shared__ float sX[64];
  __shared__ float sY[64];
  sX[t] = gx; sY[t] = gy;
  __syncthreads();
  float sumx = 0.0f, sumy = 0.0f;
  #pragma unroll
  for (int i = 0; i < 64; ++i) { sumx += sX[i]; sumy += sY[i]; }
  gX[o * 64 + t] = gx / sumx;
  gY[o * 64 + t] = gy / sumy;
}

// ---------------------------------------------------------------------------
// Pass 2: fused WMMA GEMM + gauss-weighted spatial reduction.
// Grid: (32 M-tiles, 256 N-tiles). Block: 256 threads = 8 waves (4 M x 2 N).
// Block tile: 128(O) x 128(N); wave tile 32 x 64; K double-buffered in 32-chunks.
// Per chunk each thread issues 4 b128 (A) + 16 b32 (X, transposed) = 20 async
// ops; s_wait_asynccnt 20 drains the current chunk while the next is in flight.
// ---------------------------------------------------------------------------
#define A_SZ (128 * 32)   // fp32 A chunk [o_loc][k]
#define B_SZ (128 * 36)   // fp32 X chunk [n_loc][k] (+4 pad, keeps 16B align)

__device__ __forceinline__ void load_chunk(float* Al, float* Btl,
                                           const float* __restrict__ weight,
                                           const float* __restrict__ xb,
                                           int oBase, int sBase, int kc, int t) {
  #pragma unroll
  for (int i = 0; i < 4; ++i) {      // A: 128x32 fp32, one b128 per lane
    const int idx = t + i * 256;
    const int row = idx >> 3, cg = (idx & 7) << 2;
#if HAVE_ASYNC_LDS
    async_g2l_b128(&Al[row * 32 + cg],
                   &weight[(size_t)(oBase + row) * 256 + kc + cg]);
#else
    *(float4*)&Al[row * 32 + cg] =
        *(const float4*)&weight[(size_t)(oBase + row) * 256 + kc + cg];
#endif
  }
  #pragma unroll
  for (int i = 0; i < 16; ++i) {     // X: 32x128 fp32, transposed via lane LDS addr
    const int idx = t + i * 256;
    const int cl = idx >> 7, nl = idx & 127;
#if HAVE_ASYNC_LDS
    async_g2l_b32(&Btl[nl * 36 + cl],
                  &xb[(size_t)(kc + cl) * 4096 + sBase + nl]);
#else
    Btl[nl * 36 + cl] = xb[(size_t)(kc + cl) * 4096 + sBase + nl];
#endif
  }
}

__global__ __launch_bounds__(256, 2)
void fused_gemm_kernel(const float* __restrict__ x,      // (B,C,S) fp32
                       const float* __restrict__ weight, // (O,C)   fp32
                       const float* __restrict__ gX,     // (O,64)
                       const float* __restrict__ gY,     // (O,64)
                       float* __restrict__ partial) {    // (O,512)
  __shared__ __align__(16) float Alds[2 * A_SZ];  // 32 KB double-buffered
  __shared__ __align__(16) float Bt[2 * B_SZ];    // 36 KB double-buffered

  const int t    = threadIdx.x;
  const int lane = t & 31;
  const int wave = t >> 5;
  const int wm   = wave & 3;          // 4 waves over M (32 rows each)
  const int wn   = wave >> 2;         // 2 waves over N (64 cols each)
  const int oBase = blockIdx.x * 128;
  const int nBase = blockIdx.y * 128;
  const int batch = nBase >> 12;      // n = batch*4096 + s
  const int sBase = nBase & 4095;
  const float* xb = x + (size_t)batch * (256u * 4096u);

  v8f acc[2][4] = {};

  // Prologue: stage chunk 0 into buffer 0.
  load_chunk(&Alds[0], &Bt[0], weight, xb, oBase, sBase, 0, t);

  for (int it = 0; it < 8; ++it) {            // K = 8 chunks of 32
    const int cur = it & 1;
    float* A_cur = &Alds[cur * A_SZ];
    float* B_cur = &Bt[cur * B_SZ];

    // Issue next chunk into the other buffer while this one drains.
    // (Trailing barrier of the previous iteration guarantees nobody is
    //  still reading that buffer.)
    if (it < 7)
      load_chunk(&Alds[(1 - cur) * A_SZ], &Bt[(1 - cur) * B_SZ],
                 weight, xb, oBase, sBase, (it + 1) * 32, t);

#if HAVE_ASYNC_LDS
    // Async loads retire in order: <=20 outstanding means the current
    // chunk's 20 ops have landed; the next chunk's 20 stay in flight.
    if (it < 7) {
#if __has_builtin(__builtin_amdgcn_s_wait_asynccnt)
      __builtin_amdgcn_s_wait_asynccnt(20);
#else
      asm volatile("s_wait_asynccnt 0x14" ::: "memory");
#endif
    } else {
#if __has_builtin(__builtin_amdgcn_s_wait_asynccnt)
      __builtin_amdgcn_s_wait_asynccnt(0);
#else
      asm volatile("s_wait_asynccnt 0x0" ::: "memory");
#endif
    }
#endif
    __syncthreads();   // current chunk visible to all waves

    if (it < 6) {      // warm L2 two chunks ahead -> global_prefetch_b8
      const int kp = (it + 2) * 32;
      __builtin_prefetch(&weight[(size_t)(oBase + (t >> 3)) * 256 + kp + ((t & 7) << 2)], 0, 1);
      __builtin_prefetch(&xb[(size_t)(kp + (t >> 7)) * 4096 + sBase + (t & 127)], 0, 1);
    }

    // Build fragments; A and B share contiguous k = 16*lh + s order.
    const int lr = lane & 15, lh = lane >> 4;
    v16bf aF[2], bF[4];
    #pragma unroll
    for (int mi = 0; mi < 2; ++mi) {
      // A 16x32: lane half selects K base {0,16}; contiguous 16-run
      const float* p = &A_cur[(wm * 32 + mi * 16 + lr) * 32 + lh * 16];
      aF[mi] = make_frag(p, p + 8);
    }
    #pragma unroll
    for (int ni = 0; ni < 4; ++ni) {
      // B 32x16 (stored n-major in LDS): lane half selects K base {0,16}
      const float* p = &B_cur[(wn * 64 + ni * 16 + lr) * 36 + lh * 16];
      bF[ni] = make_frag(p, p + 8);
    }
    #pragma unroll
    for (int mi = 0; mi < 2; ++mi)
      #pragma unroll
      for (int ni = 0; ni < 4; ++ni)
        acc[mi][ni] = __builtin_amdgcn_wmma_f32_16x16x32_bf16(
            false, aF[mi], false, bF[ni], (short)0, acc[mi][ni], false, false);

    __syncthreads();   // all waves done reading this buffer -> reusable
  }

  // -------- Epilogue: weight by gauss and reduce over the 64-wide n chunk ----
  const int lr = lane & 15, lh = lane >> 4;
  const int chunk = (nBase + wn * 64) >> 6;      // global 64-col chunk, 0..511
  const int iy    = (sBase >> 6) + wn;           // spatial row i = (s/64), const per wave
  #pragma unroll
  for (int mi = 0; mi < 2; ++mi) {
    float rowsum[8] = {0, 0, 0, 0, 0, 0, 0, 0};
    #pragma unroll
    for (int ni = 0; ni < 4; ++ni) {
      const int n_loc = wn * 64 + ni * 16 + lr;
      const int jx = (sBase + n_loc) & 63;       // spatial col j = s%64
      #pragma unroll
      for (int j = 0; j < 8; ++j) {
        const int o = oBase + wm * 32 + mi * 16 + j + 8 * lh;  // C layout: M=j(+8)
        float v = acc[mi][ni][j] * gX[o * 64 + jx];
        v += __shfl_xor(v, 1, 32);
        v += __shfl_xor(v, 2, 32);
        v += __shfl_xor(v, 4, 32);
        v += __shfl_xor(v, 8, 32);               // sum over 16 N lanes per half
        rowsum[j] += v;
      }
    }
    if (lr == 0) {                                // lanes 0 and 16: unique writer
      #pragma unroll
      for (int j = 0; j < 8; ++j) {
        const int o = oBase + wm * 32 + mi * 16 + j + 8 * lh;
        partial[(size_t)o * 512 + chunk] = rowsum[j] * gY[o * 64 + iy];
      }
    }
  }
}

// ---------------------------------------------------------------------------
// Pass 3: out[b,o] = sum over the 64 chunks belonging to batch b.
// ---------------------------------------------------------------------------
__global__ void reduce_kernel(const float* __restrict__ partial,
                              float* __restrict__ out) {
  const int idx = blockIdx.x * blockDim.x + threadIdx.x;  // 0..32767
  const int o = idx & 4095;
  const int b = idx >> 12;
  const float* p = partial + (size_t)o * 512 + b * 64;    // contiguous 64
  float s = 0.0f;
  #pragma unroll
  for (int j = 0; j < 64; ++j) s += p[j];
  out[(size_t)b * 4096 + o] = s;
}

extern "C" void kernel_launch(void* const* d_in, const int* in_sizes, int n_in,
                              void* d_out, int out_size, void* d_ws, size_t ws_size,
                              hipStream_t stream) {
  (void)in_sizes; (void)n_in; (void)out_size; (void)ws_size;
  const float* x      = (const float*)d_in[0];  // (8,256,64,64)
  const float* mu     = (const float*)d_in[1];  // (4096,2)
  const float* sigma  = (const float*)d_in[2];  // (4096,2)
  const float* weight = (const float*)d_in[3];  // (4096,256)

  float* gX      = (float*)d_ws;                // 4096*64
  float* gY      = gX + 4096 * 64;              // 4096*64
  float* partial = gY + 4096 * 64;              // 4096*512

  gauss_tables_kernel<<<4096, 64, 0, stream>>>(mu, sigma, gX, gY);
  dim3 grid(32, 256);                           // (O/128, N/128)
  fused_gemm_kernel<<<grid, 256, 0, stream>>>(x, weight, gX, gY, partial);
  reduce_kernel<<<128, 256, 0, stream>>>(partial, (float*)d_out);
}
